// AttentionBlock_34351148433935
// MI455X (gfx1250) — matrix operand order repaired and checked by reading, not compiled, not run on hardware
//
#include <hip/hip_runtime.h>

// ---------------------------------------------------------------------------
// Types / helpers
// ---------------------------------------------------------------------------
typedef __attribute__((ext_vector_type(16))) __bf16 v16bf;
typedef __attribute__((ext_vector_type(8)))  __bf16 v8bf;
typedef __attribute__((ext_vector_type(8)))  float  v8f;

#define DEV __device__ __forceinline__

DEV __bf16 f2bf(float f) {
  unsigned u = __builtin_bit_cast(unsigned, f);
  unsigned r = u + 0x7FFFu + ((u >> 16) & 1u);           // round-to-nearest-even
  unsigned short h = (unsigned short)(r >> 16);
  return __builtin_bit_cast(__bf16, h);
}

DEV v16bf join16(v8bf lo, v8bf hi) {
  return __builtin_shufflevector(lo, hi, 0,1,2,3,4,5,6,7,8,9,10,11,12,13,14,15);
}
// A-fragment: two separated 8-element groups (k = base+0..7 and k = base+16..23)
DEV v16bf frag_split(const __bf16* p0, const __bf16* p1) {
  return join16(*(const v8bf*)p0, *(const v8bf*)p1);
}
// B-fragment: 16 contiguous elements
DEV v16bf frag_cont(const __bf16* p) {
  return join16(*(const v8bf*)p, *(const v8bf*)(p + 8));
}
DEV v8f wmma_bf16(v16bf a, v16bf b, v8f c) {
  return __builtin_amdgcn_wmma_f32_16x16x32_bf16(false, a, false, b, (short)0, c,
                                                 false, false);
}

// Async global->LDS copy (16B per lane), tracked by ASYNCcnt.
// LDS address = low 32 bits of the generic pointer.
DEV void async_b128(__bf16* lds, const __bf16* g) {
  unsigned l = (unsigned)(size_t)(void*)lds;
  asm volatile("global_load_async_to_lds_b128 %0, %1, off"
               :: "v"(l), "v"(g) : "memory");
}
DEV void async_wait4() { asm volatile("s_wait_asynccnt 0x4" ::: "memory"); }
DEV void async_wait0() { asm volatile("s_wait_asynccnt 0x0" ::: "memory"); }

// ---------------------------------------------------------------------------
// Problem constants
// ---------------------------------------------------------------------------
#define NC   1024            // n_embed
#define NH   16              // heads
#define HD   64              // head size
#define TT   256             // sequence length
#define NB   64              // batch
#define MM   (NB * TT)       // 16384 rows

// ---------------------------------------------------------------------------
// Weight packing: fp32 [K,N] -> bf16 [N,K] (row per output column)
// ---------------------------------------------------------------------------
__global__ __launch_bounds__(256) void pack_tr_k(const float* __restrict__ src,
                                                 __bf16* __restrict__ dst,
                                                 int K, int N) {
  long long i = (long long)blockIdx.x * 256 + threadIdx.x;   // [0, K*N)
  int k = (int)(i / N);
  int n = (int)(i % N);
  dst[(size_t)n * K + k] = f2bf(src[i]);
}

// QKV weights [H,C,D] x3 -> packed bf16 [3072, C]; column j = mat*1024 + h*64 + d
__global__ __launch_bounds__(256) void pack_qkv_k(const float* __restrict__ Wq,
                                                  const float* __restrict__ Wk,
                                                  const float* __restrict__ Wv,
                                                  __bf16* __restrict__ dst) {
  int i   = blockIdx.x * 256 + threadIdx.x;  // [0, 3072*1024)
  int j   = i >> 10;
  int c   = i & 1023;
  int mat = j >> 10;
  int jj  = j & 1023;
  int hh  = jj >> 6;
  int d   = jj & 63;
  const float* W = (mat == 0) ? Wq : (mat == 1) ? Wk : Wv;
  dst[i] = f2bf(W[(size_t)((hh << 10) + c) * HD + d]);
}

// ---------------------------------------------------------------------------
// LayerNorm (fp32 in) -> bf16 out.  One row (C=1024) per 256-thread block.
// ---------------------------------------------------------------------------
__global__ __launch_bounds__(256) void ln_k(const float* __restrict__ x,
                                            const float* __restrict__ g,
                                            const float* __restrict__ be,
                                            __bf16* __restrict__ out) {
  int row = blockIdx.x, tid = threadIdx.x;
  __shared__ float rs[256], rq[256];
  float4 v = ((const float4*)x)[(size_t)row * 256 + tid];
  rs[tid] = v.x + v.y + v.z + v.w;
  rq[tid] = v.x * v.x + v.y * v.y + v.z * v.z + v.w * v.w;
  __syncthreads();
  for (int off = 128; off > 0; off >>= 1) {
    if (tid < off) { rs[tid] += rs[tid + off]; rq[tid] += rq[tid + off]; }
    __syncthreads();
  }
  float mean = rs[0] * (1.0f / NC);
  float var  = rq[0] * (1.0f / NC) - mean * mean;
  float inv  = rsqrtf(var + 1e-5f);
  int c = tid * 4;
  __bf16* op = out + (size_t)row * NC + c;
  op[0] = f2bf((v.x - mean) * inv * g[c + 0] + be[c + 0]);
  op[1] = f2bf((v.y - mean) * inv * g[c + 1] + be[c + 1]);
  op[2] = f2bf((v.z - mean) * inv * g[c + 2] + be[c + 2]);
  op[3] = f2bf((v.w - mean) * inv * g[c + 3] + be[c + 3]);
}

// ---------------------------------------------------------------------------
// Tiled bf16 GEMM:  A[M,K] (row-major) x Bp[N,K] (col-packed) -> out
// Block tile 128x128, K-step 32, 8 waves (2x4), wave tile 64x32.
// Double-buffered LDS filled by GLOBAL_LOAD_ASYNC_TO_LDS_B128 (ASYNCcnt):
// tile kb+1 streams in while WMMAs run on tile kb.
// EPI 0: scatter bf16 to q/k/v [B,H,T,D]      (out = qkv base)
// EPI 1: f32 out = acc + bias[n] + res[m*N+n] (residual add)
// EPI 2: bf16 out = relu(acc + bias[n])
// ---------------------------------------------------------------------------
template <int EPI>
__global__ __launch_bounds__(256) void gemm_k(const __bf16* __restrict__ A,
                                              const __bf16* __restrict__ Bp,
                                              const float* __restrict__ bias,
                                              const float* __restrict__ res,
                                              void* __restrict__ out,
                                              int M, int N, int K) {
  __shared__ __bf16 Al[2][128 * 40];
  __shared__ __bf16 Bl[2][128 * 40];
  const int tid  = threadIdx.x;
  const int lane = tid & 31, wid = tid >> 5;
  const int sub  = lane >> 4, l16 = lane & 15;   // lane-half, lane-in-half
  const int wm   = (wid >> 2) * 64;              // wave M offset (0/64)
  const int wn   = (wid & 3) * 32;               // wave N offset (0..96)
  const int MB   = blockIdx.y * 128;
  const int NBo  = blockIdx.x * 128;

  // this thread's two (row, col) chunks within the 128x32 tile
  const int r0 = tid >> 2,         c0 = (tid & 3) * 8;
  const int r1 = (tid + 256) >> 2, c1 = ((tid + 256) & 3) * 8;

  v8f acc[4][2] = {};
  const int nk = K >> 5;

  // prologue: stream tile 0 into buffer 0 (4 async ops per thread)
  async_b128(&Al[0][r0 * 40 + c0], A + (size_t)(MB + r0) * K + c0);
  async_b128(&Bl[0][r0 * 40 + c0], Bp + (size_t)(NBo + r0) * K + c0);
  async_b128(&Al[0][r1 * 40 + c1], A + (size_t)(MB + r1) * K + c1);
  async_b128(&Bl[0][r1 * 40 + c1], Bp + (size_t)(NBo + r1) * K + c1);

  for (int kb = 0; kb < nk; ++kb) {
    const int cur = kb & 1;
    const bool havenext = (kb + 1 < nk);
    if (havenext) {                       // stream next tile into other buffer
      const int ko = (kb + 1) * 32;
      async_b128(&Al[cur ^ 1][r0 * 40 + c0], A + (size_t)(MB + r0) * K + ko + c0);
      async_b128(&Bl[cur ^ 1][r0 * 40 + c0], Bp + (size_t)(NBo + r0) * K + ko + c0);
      async_b128(&Al[cur ^ 1][r1 * 40 + c1], A + (size_t)(MB + r1) * K + ko + c1);
      async_b128(&Bl[cur ^ 1][r1 * 40 + c1], Bp + (size_t)(NBo + r1) * K + ko + c1);
      async_wait4();                      // in-order: current tile's 4 ops done
    } else {
      async_wait0();
    }
    __syncthreads();                      // all waves' tile-kb data in LDS

    v16bf af[4], bfr[2];
#pragma unroll
    for (int mt = 0; mt < 4; ++mt) {
      const __bf16* p = &Al[cur][(wm + mt * 16 + l16) * 40 + sub * 16];
      af[mt] = frag_cont(p);
    }
#pragma unroll
    for (int nt = 0; nt < 2; ++nt)
      bfr[nt] = frag_cont(&Bl[cur][(wn + nt * 16 + l16) * 40 + sub * 16]);
#pragma unroll
    for (int mt = 0; mt < 4; ++mt)
#pragma unroll
      for (int nt = 0; nt < 2; ++nt)
        acc[mt][nt] = wmma_bf16(af[mt], bfr[nt], acc[mt][nt]);
    __syncthreads();                      // done reading buf cur (next iter overwrites)
  }

  // epilogue: C layout -> lane holds rows m = 8*sub + r, col n = l16
#pragma unroll
  for (int mt = 0; mt < 4; ++mt)
#pragma unroll
    for (int nt = 0; nt < 2; ++nt)
#pragma unroll
      for (int r = 0; r < 8; ++r) {
        int m = MB + wm + mt * 16 + sub * 8 + r;
        int n = NBo + wn + nt * 16 + l16;
        float val = acc[mt][nt][r];
        if constexpr (EPI == 0) {
          int mat = n >> 10, jj = n & 1023, hh = jj >> 6, d = jj & 63;
          int b = m >> 8, t = m & 255;
          ((__bf16*)out)[(size_t)mat * (NB * NH * TT * HD) +
                         (((size_t)(b * NH + hh) * TT + t) * HD + d)] = f2bf(val);
        } else if constexpr (EPI == 1) {
          ((float*)out)[(size_t)m * N + n] =
              val + bias[n] + res[(size_t)m * N + n];
        } else {
          float y = val + bias[n];
          ((__bf16*)out)[(size_t)m * N + n] = f2bf(y > 0.0f ? y : 0.0f);
        }
      }
}

// ---------------------------------------------------------------------------
// Attention: one block per (b,h). Flash-style online softmax.
// 8 waves, wave w owns query rows [32w, 32w+32); s-steps of 32 columns.
// scale = C^-0.5 = 1/32 (reference scales by n_embed, not head size).
// ---------------------------------------------------------------------------
__global__ __launch_bounds__(256) void attn_k(const __bf16* __restrict__ q,
                                              const __bf16* __restrict__ k,
                                              const __bf16* __restrict__ v,
                                              __bf16* __restrict__ attnO) {
  const int bh = blockIdx.x;
  const int b = bh >> 4, h = bh & 15;
  const int tid = threadIdx.x;
  const int lane = tid & 31, wid = tid >> 5;
  const int sub = lane >> 4, l16 = lane & 15;

  __shared__ __bf16 Ks[32 * 72];       // K tile   [s_local][72]   (pad: no conflicts)
  __shared__ __bf16 Vt[64 * 40];       // V^T tile [d][40]
  __shared__ __bf16 Ps[8][32 * 40];    // per-wave P scratch (C->A layout bounce)

  const __bf16* qb = q + (size_t)bh * TT * HD;
  const __bf16* kb = k + (size_t)bh * TT * HD;
  const __bf16* vb = v + (size_t)bh * TT * HD;

  // Q fragments for this wave, resident whole kernel: [mt][kk]
  v16bf qa[2][2];
#pragma unroll
  for (int mt = 0; mt < 2; ++mt)
#pragma unroll
    for (int kk = 0; kk < 2; ++kk) {
      int t = wid * 32 + mt * 16 + l16;
      const __bf16* p = qb + (size_t)t * HD + kk * 32 + sub * 16;
      qa[mt][kk] = frag_cont(p);
    }

  v8f o[2][4] = {};                    // O accumulators [mt][d-tile]
  float mrow[2][8], lrow[2][8];
#pragma unroll
  for (int mt = 0; mt < 2; ++mt)
#pragma unroll
    for (int r = 0; r < 8; ++r) { mrow[mt][r] = -3.0e38f; lrow[mt][r] = 0.0f; }

  const float scale = 0.03125f;        // 1024^-0.5

  for (int st = 0; st < 8; ++st) {
    __syncthreads();                   // previous consumers done
    {                                  // stage K rows + transposed V rows
      int s  = tid >> 3;               // 0..31
      int d0 = (tid & 7) * 8;          // 0..56
      v8bf kv = *(const v8bf*)(kb + (size_t)(st * 32 + s) * HD + d0);
      *(v8bf*)&Ks[s * 72 + d0] = kv;
      v8bf vv = *(const v8bf*)(vb + (size_t)(st * 32 + s) * HD + d0);
#pragma unroll
      for (int i = 0; i < 8; ++i) Vt[(d0 + i) * 40 + s] = vv[i];
    }
    __syncthreads();
    if (st > wid) continue;            // causal: this wave needs no later tiles

    // ---- S = scale * Q K^T  (32x32 tile) ----
    v8f sa[2][2] = {};
#pragma unroll
    for (int kk = 0; kk < 2; ++kk)
#pragma unroll
      for (int nt = 0; nt < 2; ++nt) {
        v16bf kf = frag_cont(&Ks[(nt * 16 + l16) * 72 + kk * 32 + sub * 16]);
#pragma unroll
        for (int mt = 0; mt < 2; ++mt) sa[mt][nt] = wmma_bf16(qa[mt][kk], kf, sa[mt][nt]);
      }

    // ---- mask + online softmax update ----
#pragma unroll
    for (int mt = 0; mt < 2; ++mt) {
      int rbase = wid * 32 + mt * 16 + sub * 8;
#pragma unroll
      for (int r = 0; r < 8; ++r) {
        int row = rbase + r;
        float s0 = sa[mt][0][r] * scale;
        float s1 = sa[mt][1][r] * scale;
        if (st * 32 + l16 > row)      s0 = -3.0e38f;
        if (st * 32 + 16 + l16 > row) s1 = -3.0e38f;
        float tmax = fmaxf(s0, s1);
        for (int msk = 1; msk < 16; msk <<= 1)
          tmax = fmaxf(tmax, __shfl_xor(tmax, msk, 32));
        float mold = mrow[mt][r];
        float mnew = fmaxf(mold, tmax);
        float alpha = __expf(mold - mnew);
        float p0 = __expf(s0 - mnew);
        float p1 = __expf(s1 - mnew);
        float psum = p0 + p1;
        for (int msk = 1; msk < 16; msk <<= 1)
          psum += __shfl_xor(psum, msk, 32);
        lrow[mt][r] = lrow[mt][r] * alpha + psum;
        mrow[mt][r] = mnew;
#pragma unroll
        for (int dt = 0; dt < 4; ++dt) o[mt][dt][r] *= alpha;
        int rl = mt * 16 + sub * 8 + r;           // local row in wave scratch
        Ps[wid][rl * 40 + l16]      = f2bf(p0);
        Ps[wid][rl * 40 + 16 + l16] = f2bf(p1);
      }
    }

    // ---- O += P V  (K=32) ----
#pragma unroll
    for (int mt = 0; mt < 2; ++mt) {
      const __bf16* pp = &Ps[wid][(mt * 16 + l16) * 40 + sub * 16];
      v16bf pf = frag_cont(pp);
#pragma unroll
      for (int dt = 0; dt < 4; ++dt) {
        v16bf vf = frag_cont(&Vt[(dt * 16 + l16) * 40 + sub * 16]);
        o[mt][dt] = wmma_bf16(pf, vf, o[mt][dt]);
      }
    }
  }

  // ---- normalize and scatter: attnO[b*T+t, h*64+d] ----
#pragma unroll
  for (int mt = 0; mt < 2; ++mt)
#pragma unroll
    for (int r = 0; r < 8; ++r) {
      float inv = 1.0f / lrow[mt][r];
      int t = wid * 32 + mt * 16 + sub * 8 + r;
      __bf16* op = attnO + ((size_t)(b * TT + t)) * NC + h * HD;
#pragma unroll
      for (int dt = 0; dt < 4; ++dt)
        op[dt * 16 + l16] = f2bf(o[mt][dt][r] * inv);
    }
}

// ---------------------------------------------------------------------------
// Workspace layout (bytes). f (FFN intermediate, 134,217,728 B) exactly reuses
// the dead qkv (100,663,296) + attn (33,554,432) regions. Total ~260 MB.
// ---------------------------------------------------------------------------
static const size_t OFF_WQKV  = 0;
static const size_t OFF_WPROJ = OFF_WQKV  + 3072ull * 1024 * 2;   //   6,291,456
static const size_t OFF_W1    = OFF_WPROJ + 1024ull * 1024 * 2;   //   8,388,608
static const size_t OFF_W2    = OFF_W1    + 4096ull * 1024 * 2;   //  16,777,216
static const size_t OFF_HB    = OFF_W2    + 1024ull * 4096 * 2;   //  25,165,824
static const size_t OFF_QKV   = OFF_HB    + 16384ull * 1024 * 2;  //  58,720,256
static const size_t OFF_ATTN  = OFF_QKV   + 3ull * 16777216 * 2;  // 159,383,552
static const size_t OFF_X2    = OFF_ATTN  + 16384ull * 1024 * 2;  // 192,937,984
static const size_t OFF_F     = OFF_QKV;                          // overlaps qkv+attn

extern "C" void kernel_launch(void* const* d_in, const int* in_sizes, int n_in,
                              void* d_out, int out_size, void* d_ws, size_t ws_size,
                              hipStream_t stream) {
  (void)in_sizes; (void)n_in; (void)out_size; (void)ws_size;
  const float* x     = (const float*)d_in[0];
  const float* Wq    = (const float*)d_in[1];
  const float* Wk    = (const float*)d_in[2];
  const float* Wv    = (const float*)d_in[3];
  const float* Wproj = (const float*)d_in[4];
  const float* bproj = (const float*)d_in[5];
  const float* W1    = (const float*)d_in[6];
  const float* b1    = (const float*)d_in[7];
  const float* W2    = (const float*)d_in[8];
  const float* b2    = (const float*)d_in[9];
  const float* ln1g  = (const float*)d_in[10];
  const float* ln1b  = (const float*)d_in[11];
  const float* ln2g  = (const float*)d_in[12];
  const float* ln2b  = (const float*)d_in[13];

  char* ws = (char*)d_ws;
  __bf16* wqkvP  = (__bf16*)(ws + OFF_WQKV);
  __bf16* wprojP = (__bf16*)(ws + OFF_WPROJ);
  __bf16* w1P    = (__bf16*)(ws + OFF_W1);
  __bf16* w2P    = (__bf16*)(ws + OFF_W2);
  __bf16* hb     = (__bf16*)(ws + OFF_HB);
  __bf16* qkv    = (__bf16*)(ws + OFF_QKV);
  __bf16* attnO  = (__bf16*)(ws + OFF_ATTN);
  float*  x2     = (float*) (ws + OFF_X2);
  __bf16* fbuf   = (__bf16*)(ws + OFF_F);

  // --- weight packing (bf16, [N,K]) ---
  pack_qkv_k<<<(3072 * 1024) / 256, 256, 0, stream>>>(Wq, Wk, Wv, wqkvP);
  pack_tr_k <<<(1024 * 1024) / 256, 256, 0, stream>>>(Wproj, wprojP, 1024, 1024);
  pack_tr_k <<<(1024 * 4096) / 256, 256, 0, stream>>>(W1, w1P, 1024, 4096);
  pack_tr_k <<<(4096 * 1024) / 256, 256, 0, stream>>>(W2, w2P, 4096, 1024);

  // --- attention sub-block ---
  ln_k<<<MM, 256, 0, stream>>>(x, ln1g, ln1b, hb);
  gemm_k<0><<<dim3(3072 / 128, MM / 128), 256, 0, stream>>>(
      hb, wqkvP, nullptr, nullptr, (void*)qkv, MM, 3072, 1024);
  attn_k<<<NB * NH, 256, 0, stream>>>(qkv, qkv + 16777216, qkv + 2 * 16777216, attnO);
  gemm_k<1><<<dim3(1024 / 128, MM / 128), 256, 0, stream>>>(
      attnO, wprojP, bproj, x, (void*)x2, MM, 1024, 1024);

  // --- feed-forward sub-block ---
  ln_k<<<MM, 256, 0, stream>>>(x2, ln2g, ln2b, hb);          // hb reused as h2
  gemm_k<2><<<dim3(4096 / 128, MM / 128), 256, 0, stream>>>(
      hb, w1P, b1, nullptr, (void*)fbuf, MM, 4096, 1024);
  gemm_k<1><<<dim3(1024 / 128, MM / 128), 256, 0, stream>>>(
      fbuf, w2P, b2, x2, d_out, MM, 1024, 4096);
}
